// PrototypeBank_90082644066738
// MI455X (gfx1250) — matrix operand order repaired and checked by reading, not compiled, not run on hardware
//
#include <hip/hip_runtime.h>
#include <hip/hip_bf16.h>

#define FEAT_D   1024
#define NUM_C    1000
#define KPROTO   5
#define CKTOT    (NUM_C * KPROTO)   // 5000 prototypes
#define NPAD     5120               // CKTOT padded to a multiple of BN
#define BATCH_N  16384
#define BM       64                 // rows per workgroup (2 wave-rows of 32)
#define BN       256                // proto columns per step (4 wave-cols of 64)

typedef __attribute__((ext_vector_type(16))) __bf16 v16bf;
typedef __attribute__((ext_vector_type(8)))  float  v8f;

union FragAB { uint4 q[2]; v16bf v; };

__device__ __forceinline__ unsigned int pack_bf16x2(float lo, float hi) {
  unsigned int ul = __float_as_uint(lo);
  unsigned int uh = __float_as_uint(hi);
  ul = (ul + 0x7FFFu + ((ul >> 16) & 1u)) >> 16;   // round-to-nearest-even
  uh = (uh + 0x7FFFu + ((uh >> 16) & 1u)) >> 16;
  return ul | (uh << 16);
}

// One wave32 per row: L2-normalize a 1024-float row, emit bf16.
// Rows in [rows, rowsPad) are written as zeros (tile padding for the GEMM).
__global__ void __launch_bounds__(256)
row_l2norm_to_bf16(const float* __restrict__ in, unsigned short* __restrict__ out,
                   int rows, int rowsPad) {
  const int row  = blockIdx.x * 8 + (threadIdx.x >> 5);
  const int lane = threadIdx.x & 31;
  if (row >= rowsPad) return;

  uint2* o2 = (uint2*)(out + (size_t)row * FEAT_D);   // 4 bf16 per store
  if (row >= rows) {                                  // zero padding rows
    uint2 zz; zz.x = 0u; zz.y = 0u;
    #pragma unroll
    for (int i = 0; i < 8; ++i) o2[i * 32 + lane] = zz;
    return;
  }

  const float4* r4 = (const float4*)(in + (size_t)row * FEAT_D);
  float4 v[8];
  float ss = 0.0f;
  #pragma unroll
  for (int i = 0; i < 8; ++i) {
    v[i] = r4[i * 32 + lane];
    ss += v[i].x * v[i].x + v[i].y * v[i].y + v[i].z * v[i].z + v[i].w * v[i].w;
  }
  #pragma unroll
  for (int m = 16; m >= 1; m >>= 1) ss += __shfl_xor(ss, m, 32);
  const float scale = 1.0f / fmaxf(sqrtf(ss), 1e-12f);

  #pragma unroll
  for (int i = 0; i < 8; ++i) {
    uint2 p;
    p.x = pack_bf16x2(v[i].x * scale, v[i].y * scale);
    p.y = pack_bf16x2(v[i].z * scale, v[i].w * scale);
    o2[i * 32 + lane] = p;
  }
}

// Fused GEMM (bf16 WMMA, f32 accum) + class-masked running max.
// Each wave owns a 32x64 output block (2 row-tiles x 4 col-tiles = 8
// accumulators -> 8 WMMAs per 2 A-frag + 4 B-frag loads). 8 waves tile the
// 64-row block as 2 (rows) x 4 (cols). out[0..B)=pos_sim, out[B..2B)=neg_sim.
__global__ void __launch_bounds__(256)
proto_sim_kernel(const unsigned short* __restrict__ zb,
                 const unsigned short* __restrict__ Pb,
                 const int* __restrict__ y,
                 float* __restrict__ out) {
  extern __shared__ unsigned short As[];   // BM x FEAT_D bf16 = 128 KB

  const int tid  = threadIdx.x;
  const int wave = tid >> 5;               // 0..7
  const int lane = tid & 31;
  const int m0   = blockIdx.x * BM;

  // Stage the 64x1024 bf16 A slab into LDS with 128-bit copies.
  {
    const uint4* src = (const uint4*)(zb + (size_t)m0 * FEAT_D);
    uint4* dst = (uint4*)As;
    for (int i = tid; i < BM * FEAT_D / 8; i += 256) dst[i] = src[i];
  }
  __syncthreads();

  const int wrow   = wave >> 2;            // 0..1 -> 32-row band
  const int wcol   = wave & 3;             // 0..3 -> 64-col band within BN strip
  const int half16 = lane >> 4;            // lane group
  const int l15    = lane & 15;

  // ISA 16-bit A layout: lane row = lane%16; K chunks {kb..kb+7, kb+16..kb+23}
  const int kb  = half16 * 8;
  // ISA 16-bit B layout: lane col = lane%16; K chunks {kfs..kfs+7, kfs+16..kfs+23}
  const int kfs = half16 * 8;

  const unsigned short* Abase0 = As + (wrow * 32 + l15) * FEAT_D + kb;
  const unsigned short* Abase1 = Abase0 + 16 * FEAT_D;

  // Labels for the 16 output rows this lane's accumulators cover
  // (D layout: VGPR r -> row r + half16*8; lane%16 -> column).
  int yr[16];
  #pragma unroll
  for (int i = 0; i < 2; ++i)
    #pragma unroll
    for (int r = 0; r < 8; ++r)
      yr[i * 8 + r] = y[m0 + wrow * 32 + i * 16 + half16 * 8 + r];

  float pos[16], neg[16];
  #pragma unroll
  for (int r = 0; r < 16; ++r) { pos[r] = -__builtin_inff(); neg[r] = -__builtin_inff(); }

  for (int p0 = 0; p0 < NPAD; p0 += BN) {
    const int pcol0 = p0 + wcol * 64 + l15;      // lane's proto col, col-tile 0
    const unsigned short* Bbase[4];
    #pragma unroll
    for (int j = 0; j < 4; ++j)
      Bbase[j] = Pb + (size_t)(pcol0 + j * 16) * FEAT_D + kfs;

    v8f c[2][4];
    #pragma unroll
    for (int i = 0; i < 2; ++i)
      #pragma unroll
      for (int j = 0; j < 4; ++j) c[i][j] = (v8f){};

    #pragma unroll 2
    for (int k0 = 0; k0 < FEAT_D; k0 += 32) {
      FragAB a0, a1, b[4];
      a0.q[0] = *(const uint4*)(Abase0 + k0);
      a0.q[1] = *(const uint4*)(Abase0 + k0 + 16);
      a1.q[0] = *(const uint4*)(Abase1 + k0);
      a1.q[1] = *(const uint4*)(Abase1 + k0 + 16);
      #pragma unroll
      for (int j = 0; j < 4; ++j) {
        b[j].q[0] = *(const uint4*)(Bbase[j] + k0);
        b[j].q[1] = *(const uint4*)(Bbase[j] + k0 + 16);
      }
      #pragma unroll
      for (int j = 0; j < 4; ++j) {
        c[0][j] = __builtin_amdgcn_wmma_f32_16x16x32_bf16(false, a0.v, false, b[j].v,
                                                          (short)0, c[0][j], false, false);
        c[1][j] = __builtin_amdgcn_wmma_f32_16x16x32_bf16(false, a1.v, false, b[j].v,
                                                          (short)0, c[1][j], false, false);
      }
    }

    // Masked running max (padded cols p >= CKTOT excluded entirely).
    #pragma unroll
    for (int j = 0; j < 4; ++j) {
      const int p = pcol0 + j * 16;
      if (p < CKTOT) {
        const int cls = p / KPROTO;
        #pragma unroll
        for (int r = 0; r < 8; ++r) {
          const float s0 = c[0][j][r], s1 = c[1][j][r];
          if (cls == yr[r])     pos[r]     = fmaxf(pos[r], s0);
          else                  neg[r]     = fmaxf(neg[r], s0);
          if (cls == yr[8 + r]) pos[8 + r] = fmaxf(pos[8 + r], s1);
          else                  neg[8 + r] = fmaxf(neg[8 + r], s1);
        }
      }
    }
  }

  // Max over the 16 columns of each col-tile (xor within 16-lane groups).
  #pragma unroll
  for (int m = 8; m >= 1; m >>= 1) {
    #pragma unroll
    for (int r = 0; r < 16; ++r) {
      pos[r] = fmaxf(pos[r], __shfl_xor(pos[r], m, 32));
      neg[r] = fmaxf(neg[r], __shfl_xor(neg[r], m, 32));
    }
  }

  // Merge the 4 column-band waves of each row through LDS, then store.
  __syncthreads();
  float* red = (float*)As;                  // reuse staging LDS (512 floats)
  if (l15 == 0) {
    #pragma unroll
    for (int i = 0; i < 2; ++i)
      #pragma unroll
      for (int r = 0; r < 8; ++r) {
        const int lr = wrow * 32 + i * 16 + half16 * 8 + r;   // row within block
        red[wcol * BM + lr]          = pos[i * 8 + r];
        red[4 * BM + wcol * BM + lr] = neg[i * 8 + r];
      }
  }
  __syncthreads();
  if (tid < BM) {
    float pmax = red[tid], nmax = red[4 * BM + tid];
    #pragma unroll
    for (int w = 1; w < 4; ++w) {
      pmax = fmaxf(pmax, red[w * BM + tid]);
      nmax = fmaxf(nmax, red[4 * BM + w * BM + tid]);
    }
    out[m0 + tid]           = pmax;
    out[BATCH_N + m0 + tid] = nmax;
  }
}

extern "C" void kernel_launch(void* const* d_in, const int* in_sizes, int n_in,
                              void* d_out, int out_size, void* d_ws, size_t ws_size,
                              hipStream_t stream) {
  (void)in_sizes; (void)n_in; (void)out_size; (void)ws_size;

  const float* z = (const float*)d_in[0];   // [16384, 1024] f32
  const int*   y = (const int*)d_in[1];     // [16384] labels
  const float* P = (const float*)d_in[2];   // [1000, 5, 1024] f32 = [5000, 1024]
  float* out = (float*)d_out;               // [2 * 16384] f32 (pos, neg)

  unsigned short* zb = (unsigned short*)d_ws;                    // 32 MB bf16 z_n
  unsigned short* Pb = zb + (size_t)BATCH_N * FEAT_D;            // 10.5 MB bf16 P_n (padded)

  row_l2norm_to_bf16<<<BATCH_N / 8, 256, 0, stream>>>(z, zb, BATCH_N, BATCH_N);
  row_l2norm_to_bf16<<<NPAD / 8,    256, 0, stream>>>(P, Pb, CKTOT, NPAD);

  proto_sim_kernel<<<BATCH_N / BM, 256, BM * FEAT_D * 2, stream>>>(zb, Pb, y, out);
}
